// ConditionalAffineCoupling_38010460569904
// MI455X (gfx1250) — matrix operand order repaired and checked by reading, not compiled, not run on hardware
//
#include <hip/hip_runtime.h>
#include <hip/hip_bf16.h>

typedef __bf16 bf16;
typedef __attribute__((ext_vector_type(16))) __bf16        v16bf;
typedef __attribute__((ext_vector_type(8)))  float         v8f;
typedef __attribute__((ext_vector_type(4)))  unsigned int  v4u;
typedef __attribute__((ext_vector_type(8)))  unsigned int  v8u;

// Problem constants (from reference)
constexpr int BATCH = 131072;
constexpr int DIM   = 512;
constexpr int CTX   = 128;
constexpr int HID   = 2048;
constexpr int DM    = 256;          // masked half = DIM/2
constexpr int K1    = DM + CTX;     // 384

// Tiling
constexpr int MT    = 64;           // rows per block
constexpr int LD1   = 392;          // stIn LDS stride (bf16), 384 + 8 pad
constexpr int LDH   = 2056;         // h LDS stride (bf16), 2048 + 8 pad
constexpr int LDS2  = 516;          // st LDS stride (f32), 512 + 4 pad
constexpr int STIN_BYTES = MT * LD1 * 2;              // 50176
constexpr int H_BYTES    = MT * LDH * 2;              // 263168
constexpr int LDS_BYTES  = STIN_BYTES + H_BYTES;      // 313344 (< 320KB WGP)

constexpr int W1_ELEMS = K1 * HID;        // 786432
constexpr int W2_ELEMS = HID * DIM;       // 1048576
constexpr size_t W1P_BYTES = (size_t)W1_ELEMS * 2;    // ws offset for w2p

// ---------------------------------------------------------------------------
// Prep: f32 -> bf16 weight conversion + pack into WMMA B-fragment lane order.
// B fragment (32x16, bf16, 8 VGPR / v16bf per lane):
//   lane L in [0,16):  N = L,     K = {0..7, 16..23}   (e = 0..15)
//   lane L in [16,32): N = L-16,  K = {8..15, 24..31}
// Fragment storage: frag*512 + lane*16 + e  (each lane = one 32B load).
// W1 frags ordered [nb][kb] (kb inner = GEMM1 K-loop streaming).
// W2 frags ordered [kb][nb] (nb inner = GEMM2 n-loop streaming).
// ---------------------------------------------------------------------------
__global__ void pack_weights_kernel(const float* __restrict__ W1,
                                    const float* __restrict__ W2,
                                    bf16* __restrict__ w1p,
                                    bf16* __restrict__ w2p) {
  int idx = blockIdx.x * 256 + threadIdx.x;
  if (idx < W1_ELEMS) {
    int frag = idx >> 9, r = idx & 511;
    int lane = r >> 4, e = r & 15;
    int nb = frag / 12, kb = frag - nb * 12;
    int n = nb * 16 + (lane & 15);
    int k = kb * 32 + ((lane >> 4) << 3) + (e & 7) + ((e >> 3) << 4);
    w1p[idx] = (bf16)W1[k * HID + n];
  } else {
    int i = idx - W1_ELEMS;
    if (i < W2_ELEMS) {
      int frag = i >> 9, r = i & 511;
      int lane = r >> 4, e = r & 15;
      int kb = frag >> 5, nb = frag & 31;
      int n = nb * 16 + (lane & 15);
      int k = kb * 32 + ((lane >> 4) << 3) + (e & 7) + ((e >> 3) << 4);
      w2p[i] = (bf16)W2[k * DIM + n];
    }
  }
}

// A fragment (16x32 bf16) from a row-major LDS tile.
// lane L in [0,16):  row = L,    K elems {k0..k0+7, k0+16..k0+23}
// lane L in [16,32): row = L-16, K elems {k0+8..k0+15, k0+24..k0+31}
__device__ __forceinline__ v16bf load_a_frag(const bf16* rowp, int k0, int off8) {
  v4u lo = *(const v4u*)(rowp + k0 + off8);
  v4u hi = *(const v4u*)(rowp + k0 + 16 + off8);
  v8u c;
  c[0] = lo[0]; c[1] = lo[1]; c[2] = lo[2]; c[3] = lo[3];
  c[4] = hi[0]; c[5] = hi[1]; c[6] = hi[2]; c[7] = hi[3];
  return __builtin_bit_cast(v16bf, c);
}

__global__ __launch_bounds__(256, 1)
void coupling_kernel(const float* __restrict__ x,
                     const float* __restrict__ ctx,
                     const float* __restrict__ b1,
                     const float* __restrict__ b2,
                     const bf16* __restrict__ w1p,
                     const bf16* __restrict__ w2p,
                     float* __restrict__ y,
                     float* __restrict__ logdet) {
  extern __shared__ char smem[];
  bf16*  stIn  = (bf16*)smem;                    // 64 x 384 (stride 392) bf16
  bf16*  hS    = (bf16*)(smem + STIN_BYTES);     // 64 x 2048 (stride 2056) bf16
  float* stF   = (float*)(smem + STIN_BYTES);    // overlays hS after GEMM2
  float* parts = (float*)smem;                   // overlays stIn in epilogue

  const int tid   = threadIdx.x;
  const int lane  = tid & 31;
  const int wid   = tid >> 5;
  const int wm    = wid & 3;        // wave row group: rows wm*16..+15
  const int wn    = wid >> 2;       // wave col group
  const int mrow  = lane & 15;
  const int off8  = (lane >> 4) << 3;   // 0 for lanes 0-15, 8 for 16-31
  const int rBase = blockIdx.x * MT;

  // ---- stage st_in = [x_even | context] as bf16 into LDS ----
  for (int i = tid; i < MT * DM; i += 256) {
    int r = i >> 8, k = i & 255;
    stIn[r * LD1 + k] = (bf16)x[(size_t)(rBase + r) * DIM + 2 * k];
  }
  for (int i = tid; i < MT * CTX; i += 256) {
    int r = i >> 7, c = i & 127;
    stIn[r * LD1 + DM + c] = (bf16)ctx[(size_t)(rBase + r) * CTX + c];
  }
  __syncthreads();

  // ---- GEMM1: (64x384) @ (384x2048) -> relu -> h (bf16, LDS) ----
  const bf16* aRow1 = stIn + (wm * 16 + mrow) * LD1;
  #pragma unroll 1
  for (int nbL = 0; nbL < 64; ++nbL) {
    int nb = wn * 64 + nbL;
    v8f acc = (v8f)0.0f;
    const bf16* bp = w1p + (size_t)(nb * 12) * 512 + lane * 16;
    #pragma unroll
    for (int kb = 0; kb < 12; ++kb) {
      v16bf a = load_a_frag(aRow1, kb * 32, off8);
      v16bf b = __builtin_bit_cast(v16bf, *(const v8u*)(bp + kb * 512));
      acc = __builtin_amdgcn_wmma_f32_16x16x32_bf16(false, a, false, b,
                                                    (short)0, acc, false, false);
    }
    int col = nb * 16 + mrow;
    float bias = b1[col];
    #pragma unroll
    for (int j = 0; j < 8; ++j) {             // C layout: M = j + off8, N = mrow
      float v = acc[j] + bias;
      v = v > 0.0f ? v : 0.0f;                // ReLU
      hS[(wm * 16 + j + off8) * LDH + col] = (bf16)v;
    }
  }
  __syncthreads();

  // ---- GEMM2: (64x2048) @ (2048x512). 16 n-chunks of accumulators/wave ----
  v8f acc2[16];
  #pragma unroll
  for (int i = 0; i < 16; ++i) acc2[i] = (v8f)0.0f;
  const bf16* aRow2 = hS + (wm * 16 + mrow) * LDH;
  #pragma unroll 1
  for (int kb = 0; kb < 64; ++kb) {
    v16bf a = load_a_frag(aRow2, kb * 32, off8);   // 1 A-frag reused 16x
    const bf16* bp = w2p + ((size_t)kb * 32 + wn * 16) * 512 + lane * 16;
    #pragma unroll
    for (int i = 0; i < 16; ++i) {
      v16bf b = __builtin_bit_cast(v16bf, *(const v8u*)(bp + i * 512));
      acc2[i] = __builtin_amdgcn_wmma_f32_16x16x32_bf16(false, a, false, b,
                                                        (short)0, acc2[i], false, false);
    }
  }
  __syncthreads();   // all h reads done; safe to overlay stF on h region

  #pragma unroll
  for (int i = 0; i < 16; ++i) {
    int col = wn * 256 + i * 16 + mrow;
    float bias = b2[col];
    #pragma unroll
    for (int j = 0; j < 8; ++j)
      stF[(wm * 16 + j + off8) * LDS2 + col] = acc2[i][j] + bias;
  }
  __syncthreads();

  // ---- epilogue: s=tanh(.)*5, y_odd = x_odd*exp(s)+t, y_even = x_even,
  //      log_det[row] = sum_j s ----
  const int jc = tid;                 // 0..255, one s/t column pair per thread
  for (int r = 0; r < MT; ++r) {
    float sraw = stF[r * LDS2 + jc];
    float tv   = stF[r * LDS2 + DM + jc];
    float s    = tanhf(sraw) * 5.0f;
    float e    = __expf(s);
    size_t go  = (size_t)(rBase + r) * DIM + 2 * jc;
    float xe = x[go], xo = x[go + 1];
    y[go]     = xe;
    y[go + 1] = fmaf(xo, e, tv);
    float ssum = s;                   // wave32 reduction
    #pragma unroll
    for (int o = 16; o > 0; o >>= 1) ssum += __shfl_xor(ssum, o, 32);
    if (lane == 0) parts[r * 8 + wid] = ssum;
  }
  __syncthreads();
  if (tid < MT) {
    float acc = 0.0f;
    #pragma unroll
    for (int w = 0; w < 8; ++w) acc += parts[tid * 8 + w];
    logdet[rBase + tid] = acc;
  }
}

extern "C" void kernel_launch(void* const* d_in, const int* in_sizes, int n_in,
                              void* d_out, int out_size, void* d_ws, size_t ws_size,
                              hipStream_t stream) {
  const float* x   = (const float*)d_in[0];
  const float* ctx = (const float*)d_in[1];
  const float* W1  = (const float*)d_in[2];
  const float* b1  = (const float*)d_in[3];
  const float* W2  = (const float*)d_in[4];
  const float* b2  = (const float*)d_in[5];
  float* y      = (float*)d_out;
  float* logdet = y + (size_t)BATCH * DIM;

  bf16* w1p = (bf16*)d_ws;
  bf16* w2p = (bf16*)((char*)d_ws + W1P_BYTES);

  (void)hipFuncSetAttribute((const void*)coupling_kernel,
                            hipFuncAttributeMaxDynamicSharedMemorySize, LDS_BYTES);

  int packTotal = W1_ELEMS + W2_ELEMS;
  pack_weights_kernel<<<(packTotal + 255) / 256, 256, 0, stream>>>(W1, W2, w1p, w2p);
  coupling_kernel<<<BATCH / MT, 256, LDS_BYTES, stream>>>(x, ctx, b1, b2,
                                                          w1p, w2p, y, logdet);
}
